// HGATMol_85813446574164
// MI455X (gfx1250) — compile-verified, matrix operand order checked by reading, not run on hardware
//
#include <hip/hip_runtime.h>
#include <cstddef>
#include <cstdint>

// ---------------------------------------------------------------------------
// CDNA5 (gfx1250) HGAT-Mol implementation.
// Dense GEMMs -> v_wmma_f32_16x16x32_bf16 (wave32, f32 accum). Tiles are
// staged into LDS in fragment order (global_load_b128 -> ds_store_b64), so
// each lane fetches its WMMA operand with ds_load_b128 pairs.
// Graph softmax / scatter -> atomics.
// ---------------------------------------------------------------------------

typedef __attribute__((ext_vector_type(16))) __bf16 v16bf;
typedef __attribute__((ext_vector_type(8)))  float  v8f;

#define TPB 256

// ---------------- device helpers ----------------

__device__ __forceinline__ unsigned short f2bf(float f) {
  // float -> bf16 bits, round-to-nearest-even
  unsigned int u = __float_as_uint(f);
  u += 0x7FFFu + ((u >> 16) & 1u);
  return (unsigned short)(u >> 16);
}

__device__ __forceinline__ void atomicMaxF(float* addr, float val) {
  unsigned int* ua = (unsigned int*)addr;
  unsigned int old = *ua;
  while (__uint_as_float(old) < val) {
    unsigned int prev = atomicCAS(ua, old, __float_as_uint(val));
    if (prev == old) break;
    old = prev;
  }
}

__device__ __forceinline__ float sigf(float x) { return 1.f / (1.f + expf(-x)); }

// Map K index (0..31) within a BK=32 tile to (khalf, elem) of the 16-bit
// WMMA fragment layout (ISA 7.12.2: 16-bit A 16x32 / mirrored B).
// Property: 4-aligned k groups map to one lane and 4 contiguous elems.
__device__ __forceinline__ void kmap(int k, int& kh, int& elem) {
  int ii;
  if (k < 16) { kh = k >> 3; ii = (k & 7) >> 1; }
  else        { int k2 = k - 16; kh = k2 >> 3; ii = 4 + ((k2 & 7) >> 1); }
  elem = 2 * ii + (k & 1);
}

// ---------------- WMMA GEMM ----------------
// C[M,N] = A[M,K] @ B + (Cin) + bias.  B row-major [K,N] (transB=0) or [N,K] (transB=1).
// Block tile 64(M) x 64(N) x 32(K); 8 wave32 waves (4x2); each wave: one A
// fragment x two B fragments -> 2 WMMAs per K-step. LDS tiles are stored in
// fragment order: [band][lane][16 bf16] -> one 32B contiguous chunk per lane.

#define BM 64
#define BN 64
#define BK 32

__global__ __launch_bounds__(256) void wmma_gemm_kernel(
    const float* __restrict__ A, const float* __restrict__ Bm,
    const float* __restrict__ bias, const float* __restrict__ Cin,
    float* __restrict__ C, int M, int N, int K, int transB)
{
  __shared__ __align__(32) unsigned short Af[4][32][16];  // 4 M-bands
  __shared__ __align__(32) unsigned short Bf[4][32][16];  // 4 N-bands

  const int t    = threadIdx.x;
  const int lane = t & 31;
  const int wave = t >> 5;        // 0..7
  const int wm   = wave >> 1;     // 0..3  (M band)
  const int wn   = wave & 1;      // 0..1  (pair of N bands)
  const int bM   = blockIdx.y * BM;
  const int bN   = blockIdx.x * BN;

  const int khalf = lane >> 4;    // C/D row-half select
  const int l16   = lane & 15;

  v8f acc0, acc1;
  #pragma unroll
  for (int r = 0; r < 8; ++r) {
    int row = bM + wm * 16 + r + 8 * khalf;
    int c0  = bN + wn * 32 + l16;
    int c1  = c0 + 16;
    float v0 = 0.f, v1 = 0.f;
    if (Cin && row < M) {
      if (c0 < N) v0 = Cin[(size_t)row * N + c0];
      if (c1 < N) v1 = Cin[(size_t)row * N + c1];
    }
    acc0[r] = v0; acc1[r] = v1;
  }

  for (int k0 = 0; k0 < K; k0 += BK) {
    // stage A tile (64x32): one float4 (global_load_b128) + one ds_store_b64
    // per group of 4 consecutive K values (all land in one lane, 4 contiguous
    // fragment elements).
    #pragma unroll
    for (int i = 0; i < 2; ++i) {
      int g  = t + i * 256;            // 0..511 groups
      int r  = g >> 3;                 // tile row 0..63
      int c4 = (g & 7) * 4;            // K group start 0..28
      int gr = bM + r;
      float4 v = make_float4(0.f, 0.f, 0.f, 0.f);
      if (gr < M) v = *(const float4*)(&A[(size_t)gr * K + k0 + c4]);
      int kh, el; kmap(c4, kh, el);
      ushort4 u;
      u.x = f2bf(v.x); u.y = f2bf(v.y); u.z = f2bf(v.z); u.w = f2bf(v.w);
      *(ushort4*)(&Af[r >> 4][(r & 15) + 16 * kh][el]) = u;
    }
    // stage B tile (32(K) x 64(N)) into fragment order
    if (transB) {
      // B is [N,K]: contiguous along K -> same float4 + ds_store_b64 trick
      #pragma unroll
      for (int i = 0; i < 2; ++i) {
        int g  = t + i * 256;
        int n  = g >> 3;               // tile col 0..63
        int k4 = (g & 7) * 4;
        int gn = bN + n;
        float4 v = make_float4(0.f, 0.f, 0.f, 0.f);
        if (gn < N) v = *(const float4*)(&Bm[(size_t)gn * K + k0 + k4]);
        int kh, el; kmap(k4, kh, el);
        ushort4 u;
        u.x = f2bf(v.x); u.y = f2bf(v.y); u.z = f2bf(v.z); u.w = f2bf(v.w);
        *(ushort4*)(&Bf[n >> 4][(n & 15) + 16 * kh][el]) = u;
      }
    } else {
      // B is [K,N]: contiguous along N -> float4 over 4 columns, 4 b16 stores
      // to adjacent lanes (same band / same elem).
      #pragma unroll
      for (int i = 0; i < 2; ++i) {
        int g  = t + i * 256;
        int r  = g >> 4;               // K row 0..31
        int n4 = (g & 15) * 4;         // col group 0..60
        int gk = k0 + r;
        int gn = bN + n4;
        float4 v = make_float4(0.f, 0.f, 0.f, 0.f);
        if (gn < N) v = *(const float4*)(&Bm[(size_t)gk * N + gn]);
        int kh, el; kmap(r, kh, el);
        int band = n4 >> 4, nn = (n4 & 15) + 16 * kh;
        Bf[band][nn + 0][el] = f2bf(v.x);
        Bf[band][nn + 1][el] = f2bf(v.y);
        Bf[band][nn + 2][el] = f2bf(v.z);
        Bf[band][nn + 3][el] = f2bf(v.w);
      }
    }
    if (k0 + BK < K) {
      // gfx1250 global_prefetch_b8 for next A tile
      __builtin_prefetch(&A[(size_t)(bM + (t >> 2)) * K + k0 + BK], 0, 1);
    }
    __syncthreads();

    // per-lane fragments are contiguous 32B chunks -> ds_load_b128 pairs
    v16bf af = *(const v16bf*)(&Af[wm][lane][0]);
    v16bf b0 = *(const v16bf*)(&Bf[wn * 2][lane][0]);
    v16bf b1 = *(const v16bf*)(&Bf[wn * 2 + 1][lane][0]);
    acc0 = __builtin_amdgcn_wmma_f32_16x16x32_bf16(
        false, af, false, b0, (short)0, acc0, false, false);
    acc1 = __builtin_amdgcn_wmma_f32_16x16x32_bf16(
        false, af, false, b1, (short)0, acc1, false, false);
    __syncthreads();
  }

  #pragma unroll
  for (int r = 0; r < 8; ++r) {
    int row = bM + wm * 16 + r + 8 * khalf;
    int c0  = bN + wn * 32 + l16;
    int c1  = c0 + 16;
    if (row < M) {
      if (c0 < N) {
        float v = acc0[r];
        if (bias) v += bias[c0];
        C[(size_t)row * N + c0] = v;
      }
      if (c1 < N) {
        float v = acc1[r];
        if (bias) v += bias[c1];
        C[(size_t)row * N + c1] = v;
      }
    }
  }
}

// ---------------- elementwise / graph kernels ----------------

__global__ void fill_kernel(float* __restrict__ p, float v, size_t n) {
  size_t i = (size_t)blockIdx.x * TPB + threadIdx.x;
  size_t stride = (size_t)gridDim.x * TPB;
  for (; i < n; i += stride) p[i] = v;
}

__global__ void elu_kernel(float* __restrict__ x, size_t n) {
  size_t i = (size_t)blockIdx.x * TPB + threadIdx.x;
  size_t stride = (size_t)gridDim.x * TPB;
  for (; i < n; i += stride) { float v = x[i]; x[i] = v > 0.f ? v : (expf(v) - 1.f); }
}

// out[i,h] = sum_j z[i, h*hid + j] * attn[h, j]    (H = 4)
__global__ void attn_dot_kernel(const float* __restrict__ z, const float* __restrict__ attn,
                                float* __restrict__ out, int n, int hid) {
  int idx = blockIdx.x * TPB + threadIdx.x;
  if (idx >= n * 4) return;
  int i = idx >> 2, h = idx & 3;
  const float* zp = z + (size_t)i * 4 * hid + (size_t)h * hid;
  const float* ap = attn + (size_t)h * hid;
  float s = 0.f;
  for (int j = 0; j < hid; ++j) s += zp[j] * ap[j];
  out[idx] = s;
}

// joint segment softmax over heterogeneous edges: pass 1 (max)
__global__ void edge_max_kernel(const float* __restrict__ el, const float* __restrict__ er,
                                const int* __restrict__ srcIdx, const int* __restrict__ dstIdx,
                                int E, float* __restrict__ mbuf) {
  int idx = blockIdx.x * TPB + threadIdx.x;
  if (idx >= E * 4) return;
  int e = idx >> 2, h = idx & 3;
  int s = srcIdx ? srcIdx[e] : e;
  int d = dstIdx ? dstIdx[e] : e;
  float x = el[d * 4 + h] + er[s * 4 + h];
  x = x > 0.f ? x : 0.2f * x;                 // leaky_relu(0.2)
  atomicMaxF(&mbuf[d * 4 + h], x);
}

// pass 2 (exp + sum), stores w
__global__ void edge_sum_kernel(const float* __restrict__ el, const float* __restrict__ er,
                                const int* __restrict__ srcIdx, const int* __restrict__ dstIdx,
                                int E, const float* __restrict__ mbuf, float* __restrict__ sbuf,
                                float* __restrict__ wbuf, int wOff) {
  int idx = blockIdx.x * TPB + threadIdx.x;
  if (idx >= E * 4) return;
  int e = idx >> 2, h = idx & 3;
  int s = srcIdx ? srcIdx[e] : e;
  int d = dstIdx ? dstIdx[e] : e;
  float x = el[d * 4 + h] + er[s * 4 + h];
  x = x > 0.f ? x : 0.2f * x;
  float w = expf(x - mbuf[d * 4 + h]);
  wbuf[(size_t)(wOff + e) * 4 + h] = w;
  atomicAdd(&sbuf[d * 4 + h], w);
}

// pass 3: out[dst, h, :] += alpha * zs[src, h, :]
__global__ void edge_scatter_kernel(const float* __restrict__ zs,
                                    const int* __restrict__ srcIdx, const int* __restrict__ dstIdx,
                                    int E, const float* __restrict__ sbuf,
                                    const float* __restrict__ wbuf, int wOff,
                                    float* __restrict__ out, int hid) {
  int idx = blockIdx.x * TPB + threadIdx.x;
  if (idx >= E * 4) return;
  int e = idx >> 2, h = idx & 3;
  int s = srcIdx ? srcIdx[e] : e;
  int d = dstIdx ? dstIdx[e] : e;
  float alpha = wbuf[(size_t)(wOff + e) * 4 + h] / sbuf[d * 4 + h];
  const float* zp = zs + (size_t)s * 4 * hid + (size_t)h * hid;
  float* op = out + (size_t)d * 4 * hid + (size_t)h * hid;
  for (int j = 0; j < hid; ++j) atomicAdd(&op[j], alpha * zp[j]);
}

// LSTM gates (torch order i,f,g,o) -> c,h update
__global__ void lstm_gate_kernel(const float* __restrict__ g, float* __restrict__ c,
                                 float* __restrict__ h, int ng, int d) {
  int idx = blockIdx.x * TPB + threadIdx.x;
  if (idx >= ng * d) return;
  int row = idx / d, col = idx % d;
  const float* gr = g + (size_t)row * 4 * d;
  float gi = gr[col], gf = gr[d + col], gg = gr[2 * d + col], go = gr[3 * d + col];
  float cn = sigf(gf) * c[idx] + sigf(gi) * tanhf(gg);
  c[idx] = cn;
  h[idx] = sigf(go) * tanhf(cn);
}

// e[i] = dot(feat[i,:], q[gid[i],:])
__global__ void s2s_dot_kernel(const float* __restrict__ feat, const float* __restrict__ q,
                               const int* __restrict__ gid, float* __restrict__ e, int n, int d) {
  int i = blockIdx.x * TPB + threadIdx.x;
  if (i >= n) return;
  const float* fp = feat + (size_t)i * d;
  const float* qp = q + (size_t)gid[i] * d;
  float s = 0.f;
  for (int j = 0; j < d; ++j) s += fp[j] * qp[j];
  e[i] = s;
}

__global__ void seg_max1_kernel(const float* __restrict__ e, const int* __restrict__ gid,
                                float* __restrict__ m, int n) {
  int i = blockIdx.x * TPB + threadIdx.x;
  if (i >= n) return;
  atomicMaxF(&m[gid[i]], e[i]);
}

__global__ void seg_sum1_kernel(const float* __restrict__ e, const int* __restrict__ gid,
                                const float* __restrict__ m, float* __restrict__ s,
                                float* __restrict__ w, int n) {
  int i = blockIdx.x * TPB + threadIdx.x;
  if (i >= n) return;
  int g = gid[i];
  float v = expf(e[i] - m[g]);
  w[i] = v;
  atomicAdd(&s[g], v);
}

__global__ void s2s_scatter_kernel(const float* __restrict__ feat, const int* __restrict__ gid,
                                   const float* __restrict__ w, const float* __restrict__ s,
                                   float* __restrict__ r, int n, int d) {
  size_t idx = (size_t)blockIdx.x * TPB + threadIdx.x;
  if (idx >= (size_t)n * d) return;
  int i = (int)(idx / d), j = (int)(idx % d);
  int g = gid[i];
  atomicAdd(&r[(size_t)g * d + j], (w[i] / s[g]) * feat[idx]);
}

__global__ void qstar_kernel(const float* __restrict__ q, const float* __restrict__ r,
                             float* __restrict__ qstar, int ng, int d) {
  size_t idx = (size_t)blockIdx.x * TPB + threadIdx.x;
  if (idx >= (size_t)ng * 2 * d) return;
  int row = (int)(idx / (2 * d)), j = (int)(idx % (2 * d));
  qstar[idx] = (j < d) ? q[(size_t)row * d + j] : r[(size_t)row * d + (j - d)];
}

__global__ void concat_head_kernel(const float* __restrict__ sa, const float* __restrict__ sb,
                                   const float* __restrict__ g, float* __restrict__ out, int ng) {
  size_t idx = (size_t)blockIdx.x * TPB + threadIdx.x;
  if (idx >= (size_t)ng * 2560) return;
  int row = (int)(idx / 2560), j = (int)(idx % 2560);
  float v;
  if (j < 1024)      v = sa[(size_t)row * 1024 + j];
  else if (j < 2048) v = sb[(size_t)row * 1024 + (j - 1024)];
  else               v = g[(size_t)row * 512 + (j - 2048)];
  out[idx] = v;
}

__global__ void gemv_kernel(const float* __restrict__ A, const float* __restrict__ w,
                            const float* __restrict__ b, float* __restrict__ out, int M, int K) {
  int i = blockIdx.x * TPB + threadIdx.x;
  if (i >= M) return;
  float s = b[0];
  for (int k = 0; k < K; ++k) s += A[(size_t)i * K + k] * w[k];
  out[i] = s;
}

// ---------------- host helpers ----------------

static inline void run_gemm(hipStream_t st, const float* A, const float* B, const float* bias,
                            const float* Cin, float* C, int M, int N, int K, int transB) {
  dim3 g((N + 63) / 64, (M + 63) / 64);
  wmma_gemm_kernel<<<g, 256, 0, st>>>(A, B, bias, Cin, C, M, N, K, transB);
}

static inline void run_fill(hipStream_t st, float* p, float v, size_t n) {
  size_t b = (n + TPB - 1) / TPB;
  int blocks = (int)(b < 4096 ? (b ? b : 1) : 4096);
  fill_kernel<<<blocks, TPB, 0, st>>>(p, v, n);
}

static inline void run_elu(hipStream_t st, float* p, size_t n) {
  size_t b = (n + TPB - 1) / TPB;
  int blocks = (int)(b < 4096 ? (b ? b : 1) : 4096);
  elu_kernel<<<blocks, TPB, 0, st>>>(p, n);
}

// ---------------- launch ----------------

extern "C" void kernel_launch(void* const* d_in, const int* in_sizes, int n_in,
                              void* d_out, int out_size, void* d_ws, size_t ws_size,
                              hipStream_t stream) {
  if (n_in < 99) return;
  const int IN_FEAT = 64, H = 4, Dmax = 512;
  const int NA = in_sizes[0] / IN_FEAT;   // 32768
  const int NB = in_sizes[1] / IN_FEAT;   // 34816
  const int NG = in_sizes[2] / IN_FEAT;   // 1024

  const float* atom_in   = (const float*)d_in[0];
  const float* bond_in   = (const float*)d_in[1];
  const float* glob_in   = (const float*)d_in[2];
  const int* a2b_src     = (const int*)d_in[3];   // [2*NB] atom ids
  const int* a2b_dst     = (const int*)d_in[4];   // [2*NB] bond ids
  const int* atom2graph  = (const int*)d_in[5];
  const int* bond2graph  = (const int*)d_in[6];

  // ---- params in jax tree_leaves (sorted-dict) order ----
  int pi = 7;
  const float *fc_b[4], *fc_w[4];
  for (int i = 0; i < 4; ++i) { fc_b[i] = (const float*)d_in[pi++]; fc_w[i] = (const float*)d_in[pi++]; }

  struct GP { const float *al, *ar0, *ar1, *fd, *fs0, *fs1, *res; } gp[3][3];
  for (int li = 0; li < 3; ++li)
    for (int m = 0; m < 3; ++m) {
      gp[li][m].al  = (const float*)d_in[pi++];
      gp[li][m].ar0 = (const float*)d_in[pi++];
      gp[li][m].ar1 = (const float*)d_in[pi++];
      gp[li][m].fd  = (const float*)d_in[pi++];
      gp[li][m].fs0 = (const float*)d_in[pi++];
      gp[li][m].fs1 = (const float*)d_in[pi++];
      gp[li][m].res = (li > 0) ? (const float*)d_in[pi++] : nullptr;
    }

  const float *laBhh[3], *laBih[3], *laWhh[3], *laWih[3];
  const float *lbBhh[3], *lbBih[3], *lbWhh[3], *lbWih[3];
  for (int i = 0; i < 3; ++i) laBhh[i] = (const float*)d_in[pi++];
  for (int i = 0; i < 3; ++i) laBih[i] = (const float*)d_in[pi++];
  for (int i = 0; i < 3; ++i) laWhh[i] = (const float*)d_in[pi++];
  for (int i = 0; i < 3; ++i) laWih[i] = (const float*)d_in[pi++];
  for (int i = 0; i < 3; ++i) lbBhh[i] = (const float*)d_in[pi++];
  for (int i = 0; i < 3; ++i) lbBih[i] = (const float*)d_in[pi++];
  for (int i = 0; i < 3; ++i) lbWhh[i] = (const float*)d_in[pi++];
  for (int i = 0; i < 3; ++i) lbWih[i] = (const float*)d_in[pi++];

  // ---- workspace bump allocator ----
  char* base = (char*)d_ws;
  size_t off = 0;
  auto alloc = [&](size_t nf) -> float* {
    float* p = (float*)(base + off);
    off += ((nf * sizeof(float) + 255) & ~(size_t)255);
    return p;
  };
  float* feat[3];
  feat[0] = alloc((size_t)NA * Dmax);
  feat[1] = alloc((size_t)NB * Dmax);
  feat[2] = alloc((size_t)NG * Dmax);
  float* outbuf = alloc((size_t)NB * Dmax);
  float* zdst   = alloc((size_t)NB * Dmax);
  float* zs0    = alloc((size_t)NB * Dmax);
  float* zs1    = alloc((size_t)NB * Dmax);
  float* el     = alloc((size_t)NB * H);
  float* er0    = alloc((size_t)NB * H);
  float* er1    = alloc((size_t)NB * H);
  float* mb     = alloc((size_t)NB * H);
  float* sb     = alloc((size_t)NB * H);
  float* wb     = alloc((size_t)3 * NB * H);
  float* qsA    = alloc((size_t)NG * 2 * Dmax);
  float* qsB    = alloc((size_t)NG * 2 * Dmax);
  float* hb[3]; float* cb[3];
  for (int l = 0; l < 3; ++l) { hb[l] = alloc((size_t)NG * Dmax); cb[l] = alloc((size_t)NG * Dmax); }
  float* gbuf = alloc((size_t)NG * 4 * Dmax);
  float* ebuf = alloc((size_t)NB);
  float* m1   = alloc((size_t)NG);
  float* s1   = alloc((size_t)NG);
  float* w1   = alloc((size_t)NB);
  float* rbuf = alloc((size_t)NG * Dmax);
  float* hcat = alloc((size_t)NG * 2560);
  float* f1   = alloc((size_t)NG * 128);
  float* f2   = alloc((size_t)NG * 64);
  float* f3   = alloc((size_t)NG * 32);
  if (off > ws_size) return;  // insufficient workspace: bail rather than corrupt

  // ---- initial features ----
  hipMemcpyAsync(feat[0], atom_in, (size_t)NA * IN_FEAT * sizeof(float), hipMemcpyDeviceToDevice, stream);
  hipMemcpyAsync(feat[1], bond_in, (size_t)NB * IN_FEAT * sizeof(float), hipMemcpyDeviceToDevice, stream);
  hipMemcpyAsync(feat[2], glob_in, (size_t)NG * IN_FEAT * sizeof(float), hipMemcpyDeviceToDevice, stream);

  // ---- HGAT layers ----
  int dims[3]  = {IN_FEAT, IN_FEAT, IN_FEAT};
  int nmArr[3] = {NA, NB, NG};
  int nbr[3][2] = {{1, 2}, {0, 2}, {0, 1}};   // neighbor type per master (sorted == NBRS order)
  struct ELt { int E; const int* src; const int* dst; };
  ELt edg[3][2] = {
    { {2 * NB, a2b_dst, a2b_src}, {NA, atom2graph, nullptr} },       // master atom
    { {2 * NB, a2b_src, a2b_dst}, {NB, bond2graph, nullptr} },       // master bond
    { {NA, nullptr, atom2graph},  {NB, nullptr, bond2graph} },       // master global
  };
  int hids[3] = {32, 64, 128};

  for (int li = 0; li < 3; ++li) {
    int hid = hids[li], N = 4 * hid;
    for (int m = 0; m < 3; ++m) {
      const GP& P = gp[li][m];
      int nm = nmArr[m];
      int n0 = nbr[m][0], n1 = nbr[m][1];

      run_gemm(stream, feat[m], P.fd, nullptr, nullptr, zdst, nm, N, dims[m], 0);
      attn_dot_kernel<<<(nm * H + TPB - 1) / TPB, TPB, 0, stream>>>(zdst, P.al, el, nm, hid);

      run_gemm(stream, feat[n0], P.fs0, nullptr, nullptr, zs0, nmArr[n0], N, dims[n0], 0);
      attn_dot_kernel<<<(nmArr[n0] * H + TPB - 1) / TPB, TPB, 0, stream>>>(zs0, P.ar0, er0, nmArr[n0], hid);

      run_gemm(stream, feat[n1], P.fs1, nullptr, nullptr, zs1, nmArr[n1], N, dims[n1], 0);
      attn_dot_kernel<<<(nmArr[n1] * H + TPB - 1) / TPB, TPB, 0, stream>>>(zs1, P.ar1, er1, nmArr[n1], hid);

      run_fill(stream, mb, -3.4e38f, (size_t)nm * H);
      run_fill(stream, sb, 0.f, (size_t)nm * H);
      for (int s = 0; s < 2; ++s) {
        const ELt& ed = edg[m][s];
        const float* er = (s == 0) ? er0 : er1;
        edge_max_kernel<<<(ed.E * H + TPB - 1) / TPB, TPB, 0, stream>>>(el, er, ed.src, ed.dst, ed.E, mb);
      }
      int wOff = 0;
      for (int s = 0; s < 2; ++s) {
        const ELt& ed = edg[m][s];
        const float* er = (s == 0) ? er0 : er1;
        edge_sum_kernel<<<(ed.E * H + TPB - 1) / TPB, TPB, 0, stream>>>(el, er, ed.src, ed.dst, ed.E, mb, sb, wb, wOff);
        wOff += ed.E;
      }
      if (P.res) run_gemm(stream, feat[m], P.res, nullptr, nullptr, outbuf, nm, N, dims[m], 0);
      else       run_fill(stream, outbuf, 0.f, (size_t)nm * N);
      wOff = 0;
      for (int s = 0; s < 2; ++s) {
        const ELt& ed = edg[m][s];
        const float* zsrc = (s == 0) ? zs0 : zs1;
        edge_scatter_kernel<<<(ed.E * H + TPB - 1) / TPB, TPB, 0, stream>>>(zsrc, ed.src, ed.dst, ed.E, sb, wb, wOff, outbuf, hid);
        wOff += ed.E;
      }
      run_elu(stream, outbuf, (size_t)nm * N);
      hipMemcpyAsync(feat[m], outbuf, (size_t)nm * N * sizeof(float), hipMemcpyDeviceToDevice, stream);
      dims[m] = N;
    }
  }

  // ---- set2set (5 iters, 3-layer LSTM, D=512) ----
  auto set2set = [&](const float* ft, const int* gid, int n,
                     const float* const* bhh, const float* const* bih,
                     const float* const* whh, const float* const* wih, float* qstar) {
    run_fill(stream, qstar, 0.f, (size_t)NG * 1024);
    for (int l = 0; l < 3; ++l) {
      run_fill(stream, hb[l], 0.f, (size_t)NG * 512);
      run_fill(stream, cb[l], 0.f, (size_t)NG * 512);
    }
    for (int it = 0; it < 5; ++it) {
      const float* x = qstar; int inD = 1024;
      for (int l = 0; l < 3; ++l) {
        run_gemm(stream, x, wih[l], bih[l], nullptr, gbuf, NG, 2048, inD, 1);          // x @ w_ih^T + b_ih
        run_gemm(stream, hb[l], whh[l], bhh[l], gbuf, gbuf, NG, 2048, 512, 1);         // + h @ w_hh^T + b_hh
        int tot = NG * 512;
        lstm_gate_kernel<<<(tot + TPB - 1) / TPB, TPB, 0, stream>>>(gbuf, cb[l], hb[l], NG, 512);
        x = hb[l]; inD = 512;
      }
      s2s_dot_kernel<<<(n + TPB - 1) / TPB, TPB, 0, stream>>>(ft, hb[2], gid, ebuf, n, 512);
      run_fill(stream, m1, -3.4e38f, (size_t)NG);
      run_fill(stream, s1, 0.f, (size_t)NG);
      seg_max1_kernel<<<(n + TPB - 1) / TPB, TPB, 0, stream>>>(ebuf, gid, m1, n);
      seg_sum1_kernel<<<(n + TPB - 1) / TPB, TPB, 0, stream>>>(ebuf, gid, m1, s1, w1, n);
      run_fill(stream, rbuf, 0.f, (size_t)NG * 512);
      size_t tot2 = (size_t)n * 512;
      s2s_scatter_kernel<<<(int)((tot2 + TPB - 1) / TPB), TPB, 0, stream>>>(ft, gid, w1, s1, rbuf, n, 512);
      size_t tq = (size_t)NG * 1024;
      qstar_kernel<<<(int)((tq + TPB - 1) / TPB), TPB, 0, stream>>>(hb[2], rbuf, qstar, NG, 512);
    }
  };

  set2set(feat[0], atom2graph, NA, laBhh, laBih, laWhh, laWih, qsA);
  set2set(feat[1], bond2graph, NB, lbBhh, lbBih, lbWhh, lbWih, qsB);

  // ---- head MLP ----
  size_t th = (size_t)NG * 2560;
  concat_head_kernel<<<(int)((th + TPB - 1) / TPB), TPB, 0, stream>>>(qsA, qsB, feat[2], hcat, NG);
  run_gemm(stream, hcat, fc_w[0], fc_b[0], nullptr, f1, NG, 128, 2560, 0);
  run_elu(stream, f1, (size_t)NG * 128);
  run_gemm(stream, f1, fc_w[1], fc_b[1], nullptr, f2, NG, 64, 128, 0);
  run_elu(stream, f2, (size_t)NG * 64);
  run_gemm(stream, f2, fc_w[2], fc_b[2], nullptr, f3, NG, 32, 64, 0);
  run_elu(stream, f3, (size_t)NG * 32);
  gemv_kernel<<<(NG + TPB - 1) / TPB, TPB, 0, stream>>>(f3, fc_w[3], fc_b[3], (float*)d_out, NG, 32);
}
